// SelfAttentionV1_68118181314923
// MI455X (gfx1250) — matrix operand order repaired and checked by reading, not compiled, not run on hardware
//
#include <hip/hip_runtime.h>

// SEQ=8192, D_IN=1024, D_OUT=1024 self-attention, fp32 in/out.
// Phase 1: Q/K/V projections via bf16 WMMA (LDS-staged, f32 accum); Q,K row-major
//          bf16, V transposed bf16 into workspace (48 MB total -> L2-resident).
// Phase 2: flash attention, 32 query rows per WG, 64 keys per iteration,
//          online softmax in f32, WMMA for both Q@K^T and P@V.

typedef __attribute__((ext_vector_type(8)))  __bf16 bf16x8;
typedef __attribute__((ext_vector_type(16))) __bf16 bf16x16;
typedef __attribute__((ext_vector_type(8)))  float  f32x8;

#define SEQ   8192
#define DIM   1024
#define SCALE 0.03125f  // 1/sqrt(1024)

static __device__ inline f32x8 wmma_bf16(bf16x16 a, bf16x16 b, f32x8 c) {
    // v_wmma_f32_16x16x32_bf16: D = A(16x32) * B(32x16) + C(16x16 f32)
    return __builtin_amdgcn_wmma_f32_16x16x32_bf16(
        false, a, false, b, (short)0, c, false, false);
}

// A-matrix 16x32 bf16 fragment from row-major [row][k] buffer.
// lane L: row = L%16, g = L/16; element e: K = 16g + e (contiguous 32B run).
static __device__ inline bf16x16 load_a_frag(const __bf16* base, int stride, int lane) {
    const int m = lane & 15, g = lane >> 4;
    const __bf16* p = base + (long)m * stride + 16 * g;
    bf16x8 lo = *(const bf16x8*)(p);       // K = 16g + 0..7
    bf16x8 hi = *(const bf16x8*)(p + 8);   // K = 16g + 8..15
    bf16x16 r;
#pragma unroll
    for (int i = 0; i < 8; ++i) { r[i] = lo[i]; r[8 + i] = hi[i]; }
    return r;
}

// B-matrix 32x16 bf16 fragment; buffer laid out as [n][k] (column n contiguous in k).
// lane L: n = L%16, g = L/16; element e: K = 16g + e (contiguous 32B run).
static __device__ inline bf16x16 load_b_frag(const __bf16* base, int stride, int lane) {
    const int n = lane & 15, g = lane >> 4;
    const __bf16* p = base + (long)n * stride + 16 * g;
    bf16x8 lo = *(const bf16x8*)(p);      // K = 16g + 0..7
    bf16x8 hi = *(const bf16x8*)(p + 8);  // K = 16g + 8..15
    bf16x16 r;
#pragma unroll
    for (int i = 0; i < 8; ++i) { r[i] = lo[i]; r[8 + i] = hi[i]; }
    return r;
}

// ---------------------------------------------------------------------------
// Phase 1: out_z = bf16( f32(x) @ W_z ).  grid = (DIM/128, SEQ/128, 3).
// Block tile 128x128, 8 waves as 4x2, each wave 32x64 (2x4 WMMA tiles), K-step 32.
// z==2 (V) is stored transposed (Vt[col][row]) via a branchless LDS repack.
// ---------------------------------------------------------------------------
__global__ __launch_bounds__(256) void qkv_gemm_kernel(
    const float* __restrict__ x,
    const float* __restrict__ Wq, const float* __restrict__ Wk,
    const float* __restrict__ Wv,
    __bf16* __restrict__ Qo, __bf16* __restrict__ Ko, __bf16* __restrict__ Vto)
{
    const int n0 = blockIdx.x * 128;
    const int m0 = blockIdx.y * 128;
    const int z  = blockIdx.z;
    const float* W  = (z == 0) ? Wq : (z == 1) ? Wk : Wv;
    __bf16* out     = (z == 0) ? Qo : (z == 1) ? Ko : Vto;

    const int t = threadIdx.x;
    const int lane = t & 31, wave = t >> 5;
    const int wrow = wave >> 1;        // 0..3 -> 32-row slab
    const int wcol = wave & 1;         // 0..1 -> 64-col slab

    // Shared pool: staging (sA 128x40 + sB 128x40 = 10240 el) then reused as the
    // 128x136 output repack tile (17408 el = 34816 B). 80B/272B strides are
    // 16B-aligned and land on conflict-free bank strides for the frag reads.
    __shared__ __align__(16) __bf16 smem[17408];
    __bf16* sA = smem;            // [m][k], stride 40
    __bf16* sB = smem + 5120;     // [n][k], stride 40 (W tile transposed)

    f32x8 acc[2][4];
#pragma unroll
    for (int i = 0; i < 2; ++i)
#pragma unroll
        for (int j = 0; j < 4; ++j)
#pragma unroll
            for (int r = 0; r < 8; ++r) acc[i][j][r] = 0.0f;

    const int ar = t >> 1, ac = (t & 1) * 16;   // A staging coords
    const int bk = t >> 3, bn = (t & 7) * 16;   // B staging coords

    for (int k0 = 0; k0 < DIM; k0 += 32) {
        { // stage A: 128x32 f32 -> bf16, vectorized ds_store_b128
            const float* src = x + (long)(m0 + ar) * DIM + k0 + ac;
            bf16x8 v0, v1;
#pragma unroll
            for (int i = 0; i < 8; ++i) { v0[i] = (__bf16)src[i]; v1[i] = (__bf16)src[8 + i]; }
            *(bf16x8*)&sA[ar * 40 + ac]     = v0;
            *(bf16x8*)&sA[ar * 40 + ac + 8] = v1;
        }
        { // stage B transposed: 32x128 f32 -> bf16 [n][k]
            const float* src = W + (long)(k0 + bk) * DIM + n0 + bn;
#pragma unroll
            for (int i = 0; i < 16; ++i) sB[(bn + i) * 40 + bk] = (__bf16)src[i];
        }
        if (k0 + 32 < DIM) { // prefetch next K-step tiles (global_prefetch_b8)
            __builtin_prefetch(x + (long)(m0 + ar) * DIM + k0 + 32 + ac, 0, 3);
            __builtin_prefetch(W + (long)(k0 + 32 + bk) * DIM + n0 + bn, 0, 3);
        }
        __syncthreads();

        bf16x16 a0 = load_a_frag(&sA[(wrow * 32 +  0) * 40], 40, lane);
        bf16x16 a1 = load_a_frag(&sA[(wrow * 32 + 16) * 40], 40, lane);
#pragma unroll
        for (int j = 0; j < 4; ++j) {
            bf16x16 b = load_b_frag(&sB[(wcol * 64 + j * 16) * 40], 40, lane);
            acc[0][j] = wmma_bf16(a0, b, acc[0][j]);
            acc[1][j] = wmma_bf16(a1, b, acc[1][j]);
        }
        __syncthreads();
    }

    // ---- branchless epilogue: accum -> LDS tile -> coalesced b128 stores ----
    const int g = lane >> 4, ln = lane & 15;
    const int rmul = (z == 2) ? 1   : 136;   // tile[row][col] or tile[col][row]
    const int cmul = (z == 2) ? 136 : 1;
#pragma unroll
    for (int i = 0; i < 2; ++i)
#pragma unroll
        for (int j = 0; j < 4; ++j)
#pragma unroll
            for (int r = 0; r < 8; ++r) {
                const int row = wrow * 32 + 16 * i + 8 * g + r;
                const int col = wcol * 64 + 16 * j + ln;
                smem[rmul * row + cmul * col] = (__bf16)acc[i][j][r];
            }
    __syncthreads();

    __bf16* obase = (z == 2) ? out + (long)n0 * SEQ + m0
                             : out + (long)m0 * DIM + n0;
    const long ostride = (z == 2) ? SEQ : DIM;
    const int cr = t >> 1, cc = (t & 1) * 64;       // 64 bf16 = 128B per thread
    const __bf16* srcp = &smem[cr * 136 + cc];
    __bf16* dstp = obase + (long)cr * ostride + cc;
#pragma unroll
    for (int i = 0; i < 8; ++i)
        *(bf16x8*)(dstp + 8 * i) = *(const bf16x8*)(srcp + 8 * i);
}

// ---------------------------------------------------------------------------
// Phase 2: flash attention. grid = SEQ/32 blocks, 256 threads (8 waves).
// Per WG: 32 query rows. Key loop step 64. Wave w computes S sub-tile
// (16si x 16sj) of the 32x64 score tile; wave 0 does online softmax; all waves
// own a disjoint 128-col slice of the 32x1024 f32 output accumulator.
// K^T and V^T B-fragments stream straight from global (L2-resident, 48 MB).
// ---------------------------------------------------------------------------
__global__ __launch_bounds__(256, 1) void flash_attn_kernel(
    const __bf16* __restrict__ Q, const __bf16* __restrict__ K,
    const __bf16* __restrict__ Vt, float* __restrict__ out)
{
    const int q0   = blockIdx.x * 32;
    const int t    = threadIdx.x;
    const int lane = t & 31, wave = t >> 5;
    const int si = wave >> 2, sj = wave & 3;       // S tile coords (2x4)
    const int g = lane >> 4, ln = lane & 15;

    __shared__ __align__(16) float  s_buf[32][68];  // 32x64 scores (f32)
    __shared__ __align__(16) __bf16 p_buf[32][72];  // 32x64 probs (bf16), 144B stride
    __shared__ float corr_buf[32];
    __shared__ float l_buf[32];

    f32x8 acc[2][8];
#pragma unroll
    for (int i = 0; i < 2; ++i)
#pragma unroll
        for (int j = 0; j < 8; ++j)
#pragma unroll
            for (int r = 0; r < 8; ++r) acc[i][j][r] = 0.0f;

    float m_run = -__builtin_inff();  // per-row state, valid in wave 0 (lane==row)
    float l_run = 0.0f;

    const __bf16* qbase = Q + (long)(q0 + 16 * si) * DIM;

    for (int key0 = 0; key0 < SEQ; key0 += 64) {
        if (key0 + 64 < SEQ) { // prefetch next key block (K row + Vt row starts)
            __builtin_prefetch(K + (long)(key0 + 64 + 16 * sj + ln) * DIM, 0, 2);
            __builtin_prefetch(Vt + (long)(wave * 128 + lane * 4) * SEQ + key0 + 64, 0, 2);
        }

        // ---- S tile: Q[16si.. , :] @ K^T[: , 16sj..] over full DIM -----------
        f32x8 s;
#pragma unroll
        for (int r = 0; r < 8; ++r) s[r] = 0.0f;
        const __bf16* kbase = K + (long)(key0 + 16 * sj) * DIM;
#pragma unroll 2
        for (int f0 = 0; f0 < DIM; f0 += 32) {
            bf16x16 a = load_a_frag(qbase + f0, DIM, lane);
            bf16x16 b = load_b_frag(kbase + f0, DIM, lane); // lane=key, e=feature
            s = wmma_bf16(a, b, s);
        }
#pragma unroll
        for (int r = 0; r < 8; ++r)
            s_buf[16 * si + 8 * g + r][16 * sj + ln] = s[r];
        __syncthreads();

        // ---- online softmax: wave 0, one lane per query row ------------------
        if (wave == 0) {
            const int row = lane;
            float mx = m_run;
#pragma unroll 4
            for (int c = 0; c < 64; ++c) mx = fmaxf(mx, s_buf[row][c] * SCALE);
            const float cr = __expf(m_run - mx);   // 0 on first iteration
            float sum = 0.0f;
#pragma unroll 4
            for (int c = 0; c < 64; ++c) {
                const float e = __expf(s_buf[row][c] * SCALE - mx);
                sum += e;
                p_buf[row][c] = (__bf16)e;
            }
            l_run = l_run * cr + sum;
            m_run = mx;
            corr_buf[row] = cr;
        }
        __syncthreads();

        // ---- rescale accumulators by per-row correction ----------------------
        float c0[8], c1[8];
#pragma unroll
        for (int r = 0; r < 8; ++r) {
            c0[r] = corr_buf[8 * g + r];
            c1[r] = corr_buf[16 + 8 * g + r];
        }
#pragma unroll
        for (int j = 0; j < 8; ++j)
#pragma unroll
            for (int r = 0; r < 8; ++r) {
                acc[0][j][r] *= c0[r];
                acc[1][j][r] *= c1[r];
            }

        // ---- O += P @ V : wave owns cols [wave*128, wave*128+128) ------------
#pragma unroll
        for (int ks = 0; ks < 2; ++ks) {
            bf16x16 pa0 = load_a_frag(&p_buf[0][32 * ks],  72, lane);
            bf16x16 pa1 = load_a_frag(&p_buf[16][32 * ks], 72, lane);
#pragma unroll
            for (int j = 0; j < 8; ++j) {
                const __bf16* vb =
                    Vt + (long)(wave * 128 + 16 * j) * SEQ + key0 + 32 * ks;
                bf16x16 b = load_b_frag(vb, SEQ, lane); // lane=col, e=key
                acc[0][j] = wmma_bf16(pa0, b, acc[0][j]);
                acc[1][j] = wmma_bf16(pa1, b, acc[1][j]);
            }
        }
        // No trailing barrier needed: next iteration's post-S barrier orders
        // p_buf/corr_buf consumption against wave 0's next writes.
    }

    if (wave == 0) l_buf[lane] = 1.0f / l_run;
    __syncthreads();

    float i0[8], i1[8];
#pragma unroll
    for (int r = 0; r < 8; ++r) {
        i0[r] = l_buf[8 * g + r];
        i1[r] = l_buf[16 + 8 * g + r];
    }
#pragma unroll
    for (int i = 0; i < 2; ++i)
#pragma unroll
        for (int j = 0; j < 8; ++j)
#pragma unroll
            for (int r = 0; r < 8; ++r) {
                const int row = q0 + 16 * i + 8 * g + r;
                const int col = wave * 128 + 16 * j + ln;
                out[(long)row * DIM + col] = acc[i][j][r] * (i ? i1[r] : i0[r]);
            }
}

// ---------------------------------------------------------------------------
extern "C" void kernel_launch(void* const* d_in, const int* in_sizes, int n_in,
                              void* d_out, int out_size, void* d_ws, size_t ws_size,
                              hipStream_t stream) {
    (void)in_sizes; (void)n_in; (void)out_size; (void)ws_size;
    const float* x  = (const float*)d_in[0];
    const float* Wq = (const float*)d_in[1];
    const float* Wk = (const float*)d_in[2];
    const float* Wv = (const float*)d_in[3];
    float* out = (float*)d_out;

    // Workspace: Q (16MB) | K (16MB) | V^T (16MB), all bf16.
    __bf16* qws  = (__bf16*)d_ws;
    __bf16* kws  = qws + (size_t)SEQ * DIM;
    __bf16* vtws = kws + (size_t)SEQ * DIM;

    dim3 g1(DIM / 128, SEQ / 128, 3);
    qkv_gemm_kernel<<<g1, 256, 0, stream>>>(x, Wq, Wk, Wv, qws, kws, vtws);

    flash_attn_kernel<<<SEQ / 32, 256, 0, stream>>>(qws, kws, vtws, out);
}
